// Attention_66125316489672
// MI455X (gfx1250) — compile-verified, matrix operand order checked by reading, not run on hardware
//
#include <hip/hip_runtime.h>
#include <hip/hip_bf16.h>

// ---- CDNA5 vector types for WMMA ----
typedef __attribute__((ext_vector_type(16))) __bf16 v16bf;
typedef __attribute__((ext_vector_type(8)))  float  v8f;
typedef __attribute__((ext_vector_type(4)))  int    v4i;

// address-space qualified pointers for async global->LDS copies
// (clang prints AS1 as __device__, AS3 as __shared__ in diagnostics; the
//  async-load builtin takes int4 AS1* / int4 AS3* style operands)
typedef __attribute__((address_space(1))) v4i as1_v4i;
typedef __attribute__((address_space(3))) v4i as3_v4i;

#if __has_builtin(__builtin_amdgcn_global_load_async_to_lds_b128) && \
    __has_builtin(__builtin_amdgcn_s_wait_asynccnt)
#define USE_ASYNC_LDS 1
#else
#define USE_ASYNC_LDS 0
#endif

#define T5      5
#define NHEADS  8
#define DH      64
#define DIM     512
#define INNER3  1536
#define BT      160
#define NTOK    257
#define NVTOK   256
#define BATCH   32
#define MV      (BT * NVTOK)   // 40960 V rows
#define MCAT    (BT * NTOK)    // 41120 concatenated rows

#define BK       64            // K step per LDS stage
#define LDSS     80            // padded row stride (160B, 32B-aligned rows)

// ---------------------------------------------------------------------------
// Split x (f32, [160][257][512]) into bf16 A rows [160][512] and V rows
// [40960][512] so both downstream GEMMs read dense row-major bf16.
// ---------------------------------------------------------------------------
__global__ void k_split_x(const float* __restrict__ x,
                          __bf16* __restrict__ Ab,
                          __bf16* __restrict__ Vb) {
  long i = (long)blockIdx.x * blockDim.x + threadIdx.x;
  const long total = (long)BT * NTOK * DIM;
  if (i >= total) return;
  int  col = (int)(i % DIM);
  long row = i / DIM;
  int  bt  = (int)(row / NTOK);
  int  j   = (int)(row % NTOK);
  __bf16 v = (__bf16)x[i];
  if (j == 0) Ab[(long)bt * DIM + col] = v;
  else        Vb[((long)bt * NVTOK + (j - 1)) * DIM + col] = v;
}

// Convert weight [K][N] f32 -> transposed bf16 [N][K] so GEMM B-tiles are
// row-major over output columns (no in-kernel transpose needed).
__global__ void k_wT_to_bf16(const float* __restrict__ W,
                             __bf16* __restrict__ Wt, int K, int N) {
  long i = (long)blockIdx.x * blockDim.x + threadIdx.x;
  if (i >= (long)K * N) return;
  int n = (int)(i / K);
  int k = (int)(i % K);
  Wt[i] = (__bf16)W[(long)k * N + n];
}

// ---------------------------------------------------------------------------
// bf16 GEMM, f32 accumulate via v_wmma_f32_16x16x32_bf16.
//   C[MxN] = A[MxK] * Bt[NxK]^T   (row-major bf16, K multiple of 64)
// Block tile 128x128, K-step 64, double-buffered LDS staged with
// global_load_async_to_lds_b128 (ASYNCcnt) when available.
// 256 threads = 8 wave32 waves in 2(M)x4(N); wave tile 64x32 = 4x2 accums.
// ---------------------------------------------------------------------------
__launch_bounds__(256)
__global__ void k_gemm_bf16(const __bf16* __restrict__ A,
                            const __bf16* __restrict__ Bt,
                            __bf16* __restrict__ Cb,
                            float* __restrict__ Cf,
                            const float* __restrict__ bias,
                            int M, int N, int K) {
  __shared__ __align__(64) __bf16 As[2][128][LDSS];
  __shared__ __align__(64) __bf16 Bs[2][128][LDSS];

  const int tid  = threadIdx.x;
  const int lane = tid & 31;
  const int wave = tid >> 5;
  const int wm   = wave >> 2;   // 0..1
  const int wn   = wave & 3;    // 0..3
  const int m0   = blockIdx.y * 128;
  const int n0   = blockIdx.x * 128;

  v8f acc[4][2];
#pragma unroll
  for (int i = 0; i < 4; ++i)
#pragma unroll
    for (int j = 0; j < 2; ++j) acc[i][j] = (v8f)0.0f;

  // cooperative staging: thread -> (row 0..127, 32-elem K segment)
  const int lrow = tid >> 1;
  const int kseg = (tid & 1) * 32;

  int garow = m0 + lrow;
  if (garow > M - 1) garow = M - 1;          // clamp for ragged M
  const __bf16* gA = A  + (long)garow * K + kseg;
  const __bf16* gB = Bt + (long)(n0 + lrow) * K + kseg;

  auto issue = [&](int buf, int k0) {
#if USE_ASYNC_LDS
#pragma unroll
    for (int e = 0; e < 4; ++e) {
      __builtin_amdgcn_global_load_async_to_lds_b128(
          (as1_v4i*)(gA + k0 + e * 8),
          (as3_v4i*)&As[buf][lrow][kseg + e * 8], 0, 0);
      __builtin_amdgcn_global_load_async_to_lds_b128(
          (as1_v4i*)(gB + k0 + e * 8),
          (as3_v4i*)&Bs[buf][lrow][kseg + e * 8], 0, 0);
    }
#else
#pragma unroll
    for (int e = 0; e < 4; ++e) {
      uint4 pa = *reinterpret_cast<const uint4*>(gA + k0 + e * 8);
      uint4 pb = *reinterpret_cast<const uint4*>(gB + k0 + e * 8);
      *reinterpret_cast<uint4*>(&As[buf][lrow][kseg + e * 8]) = pa;
      *reinterpret_cast<uint4*>(&Bs[buf][lrow][kseg + e * 8]) = pb;
    }
#endif
  };

  const int lr = lane & 15;          // row/col within 16x16 frag
  const int kh = (lane >> 4) * 16;   // K-half selector within 32-K slab

  const int nIter = K / BK;
  issue(0, 0);
  for (int it = 0; it < nIter; ++it) {
    const int cur = it & 1;
    const bool hasNext = (it + 1 < nIter);
    if (hasNext) issue(1 - cur, (it + 1) * BK);
#if USE_ASYNC_LDS
    if (hasNext) __builtin_amdgcn_s_wait_asynccnt(8);  // older 8 (cur) done
    else         __builtin_amdgcn_s_wait_asynccnt(0);
#endif
    __syncthreads();   // cur tile visible to all waves

#pragma unroll
    for (int kk = 0; kk < BK; kk += 32) {
      const int kh2 = kk + kh;
      v16bf af[4], bfv[2];
#pragma unroll
      for (int i = 0; i < 4; ++i)
        af[i] = *reinterpret_cast<const v16bf*>(&As[cur][wm * 64 + i * 16 + lr][kh2]);
#pragma unroll
      for (int j = 0; j < 2; ++j)
        bfv[j] = *reinterpret_cast<const v16bf*>(&Bs[cur][wn * 32 + j * 16 + lr][kh2]);
#pragma unroll
      for (int i = 0; i < 4; ++i)
#pragma unroll
        for (int j = 0; j < 2; ++j)
          acc[i][j] = __builtin_amdgcn_wmma_f32_16x16x32_bf16(
              false, af[i], false, bfv[j], (short)0, acc[i][j], false, false);
    }
    __syncthreads();   // done reading cur before it is overwritten
  }

  // ---- epilogue: C frag layout = VGPR r -> M = r + 8*(lane>=16), N = lane%16
  const int crow = (lane >> 4) * 8;
  const int ccol = lane & 15;
#pragma unroll
  for (int i = 0; i < 4; ++i) {
#pragma unroll
    for (int j = 0; j < 2; ++j) {
      const int col = n0 + wn * 32 + j * 16 + ccol;
#pragma unroll
      for (int r = 0; r < 8; ++r) {
        const int row = m0 + wm * 64 + i * 16 + crow + r;
        if (row < M) {
          float v = acc[i][j][r];
          if (Cf) Cf[(long)row * N + col] = v + (bias ? bias[col] : 0.0f);
          else    Cb[(long)row * N + col] = (__bf16)v;
        }
      }
    }
  }
}

// ---------------------------------------------------------------------------
// 64-wide bf16 . f32 dot using 32B vector loads.
__device__ __forceinline__ float dot64_bf(const __bf16* __restrict__ p,
                                          const float* __restrict__ q) {
  float s = 0.f;
#pragma unroll
  for (int c = 0; c < 4; ++c) {
    v16bf vb = *reinterpret_cast<const v16bf*>(p + c * 16);
#pragma unroll
    for (int e = 0; e < 16; ++e) s += (float)vb[e] * q[c * 16 + e];
  }
  return s;
}

// ---------------------------------------------------------------------------
// Fused bidirectional cross-attention per (batch b, head h).
//   av: 5 A-queries over 1280 V-keys/values (LDS softmax over 1280)
//   va: 1280 masked V-queries over 5 A-keys/values (register softmax over 5)
// Writes directly into concatenated (160,257,512) bf16 layout.
// ---------------------------------------------------------------------------
__launch_bounds__(256)
__global__ void k_attn(const __bf16* __restrict__ vqkv,  // [40960][1536] q|k|v
                       const __bf16* __restrict__ aqkv,  // [160][1536]   q|k|v
                       const float* __restrict__ mask,   // [160][256]
                       __bf16* __restrict__ cat) {       // [41120][512]
  const int b   = blockIdx.x >> 3;
  const int h   = blockIdx.x & 7;
  const int tid = threadIdx.x;

  __shared__ float qa[T5][DH], ka[T5][DH], va[T5][DH], avo[T5][DH];
  __shared__ float red[256];
  __shared__ float P[1280];

  for (int i = tid; i < T5 * DH; i += 256) {
    int t = i / DH, d = i % DH;
    long base = (long)(b * T5 + t) * INNER3 + h * DH + d;
    qa[t][d] = (float)aqkv[base];
    ka[t][d] = (float)aqkv[base + DIM];
    va[t][d] = (float)aqkv[base + 2 * DIM];
  }
  __syncthreads();

  const float scale = 0.125f;  // dh^-0.5, dh = 64

  // ---- av: A attends to V ----
  for (int t = 0; t < T5; ++t) {
    float lmax = -1e30f;
    for (int k = tid; k < 1280; k += 256) {
      int bt = b * T5 + (k % T5);
      int n  = k / T5;
      const __bf16* kp = vqkv + (long)(bt * NVTOK + n) * INNER3 + DIM + h * DH;
      float s = dot64_bf(kp, &qa[t][0]) * scale;
      P[k] = s;
      lmax = fmaxf(lmax, s);
    }
    red[tid] = lmax; __syncthreads();
    for (int off = 128; off > 0; off >>= 1) {
      if (tid < off) red[tid] = fmaxf(red[tid], red[tid + off]);
      __syncthreads();
    }
    float mx = red[0]; __syncthreads();

    float lsum = 0.f;
    for (int k = tid; k < 1280; k += 256) {
      float e = __expf(P[k] - mx);
      P[k] = e;
      lsum += e;
    }
    red[tid] = lsum; __syncthreads();
    for (int off = 128; off > 0; off >>= 1) {
      if (tid < off) red[tid] += red[tid + off];
      __syncthreads();
    }
    float inv = 1.0f / red[0]; __syncthreads();

    {  // P (1x1280) @ vv (1280x64): 64 d-lanes x 4 k-chunks
      int d = tid & 63, c = tid >> 6;
      float part = 0.f;
      for (int k = c * 320; k < (c + 1) * 320; ++k) {
        int bt = b * T5 + (k % T5);
        int n  = k / T5;
        const __bf16* vp = vqkv + (long)(bt * NVTOK + n) * INNER3 + 2 * DIM + h * DH;
        part += P[k] * (float)vp[d];
      }
      red[tid] = part; __syncthreads();
      if (c == 0)
        avo[t][d] = (red[d] + red[64 + d] + red[128 + d] + red[192 + d]) * inv;
      __syncthreads();
    }
  }

  // a_out -> position 0 of each bt row
  for (int i = tid; i < T5 * DH; i += 256) {
    int t = i / DH, d = i % DH;
    cat[((long)(b * T5 + t) * NTOK) * DIM + h * DH + d] = (__bf16)avo[t][d];
  }

  // ---- va: masked V attends to A (softmax over 5 in registers) ----
  for (int q = tid; q < 1280; q += 256) {
    int tq = q % T5, n = q / T5;
    int bt = b * T5 + tq;
    float m = mask[(long)bt * NVTOK + n];
    const __bf16* qp = vqkv + (long)(bt * NVTOK + n) * INNER3 + h * DH;

    float qreg[DH];
#pragma unroll
    for (int c = 0; c < 4; ++c) {
      v16bf vb = *reinterpret_cast<const v16bf*>(qp + c * 16);
#pragma unroll
      for (int e = 0; e < 16; ++e) qreg[c * 16 + e] = (float)vb[e];
    }

    float s[T5], mx = -1e30f;
#pragma unroll
    for (int t = 0; t < T5; ++t) {
      float a = 0.f;
#pragma unroll 8
      for (int d = 0; d < DH; ++d) a += qreg[d] * ka[t][d];
      s[t] = a * m * scale;
      mx = fmaxf(mx, s[t]);
    }
    float sum = 0.f;
#pragma unroll
    for (int t = 0; t < T5; ++t) { s[t] = __expf(s[t] - mx); sum += s[t]; }
    float inv = 1.f / sum;

    __bf16* op = cat + ((long)bt * NTOK + 1 + n) * DIM + h * DH;
#pragma unroll
    for (int c = 0; c < 4; ++c) {
      v16bf ov;
#pragma unroll
      for (int e = 0; e < 16; ++e) {
        int d = c * 16 + e;
        float o = 0.f;
#pragma unroll
        for (int t = 0; t < T5; ++t) o += s[t] * va[t][d];
        ov[e] = (__bf16)(o * inv);
      }
      *reinterpret_cast<v16bf*>(op + c * 16) = ov;
    }
  }
}

// ---------------------------------------------------------------------------
extern "C" void kernel_launch(void* const* d_in, const int* in_sizes, int n_in,
                              void* d_out, int out_size, void* d_ws, size_t ws_size,
                              hipStream_t stream) {
  (void)in_sizes; (void)n_in; (void)out_size; (void)ws_size;
  const float* x    = (const float*)d_in[0];
  const float* mask = (const float*)d_in[1];
  const float* Wv   = (const float*)d_in[2];
  const float* Wa   = (const float*)d_in[3];
  const float* Wo   = (const float*)d_in[4];
  const float* bo   = (const float*)d_in[5];
  float* out = (float*)d_out;

  // workspace carve-up (bf16 staging, ~212 MB total)
  char* ws = (char*)d_ws;
  size_t off = 0;
  auto carve = [&](size_t bytes) -> void* {
    void* p = ws + off;
    off = (off + bytes + 255) & ~(size_t)255;
    return p;
  };
  __bf16* Vb   = (__bf16*)carve((size_t)MV * DIM * 2);       // 40 MB
  __bf16* Ab   = (__bf16*)carve((size_t)BT * DIM * 2);
  __bf16* WvT  = (__bf16*)carve((size_t)DIM * INNER3 * 2);   // [1536][512]
  __bf16* WaT  = (__bf16*)carve((size_t)DIM * INNER3 * 2);   // [1536][512]
  __bf16* WoT  = (__bf16*)carve((size_t)DIM * DIM * 2);      // [512][512]
  __bf16* vqkv = (__bf16*)carve((size_t)MV * INNER3 * 2);    // 120 MB
  __bf16* aqkv = (__bf16*)carve((size_t)BT * INNER3 * 2);
  __bf16* cat  = (__bf16*)carve((size_t)MCAT * DIM * 2);     // 40 MB

  const long nx = (long)BT * NTOK * DIM;
  k_split_x<<<(unsigned)((nx + 255) / 256), 256, 0, stream>>>(x, Ab, Vb);
  k_wT_to_bf16<<<(DIM * INNER3 + 255) / 256, 256, 0, stream>>>(Wv, WvT, DIM, INNER3);
  k_wT_to_bf16<<<(DIM * INNER3 + 255) / 256, 256, 0, stream>>>(Wa, WaT, DIM, INNER3);
  k_wT_to_bf16<<<(DIM * DIM + 255) / 256, 256, 0, stream>>>(Wo, WoT, DIM, DIM);

  // V-QKV: 40960 x 1536 x 512
  k_gemm_bf16<<<dim3(INNER3 / 128, MV / 128), 256, 0, stream>>>(
      Vb, WvT, vqkv, nullptr, nullptr, MV, INNER3, DIM);
  // A-QKV: 160 x 1536 x 512
  k_gemm_bf16<<<dim3(INNER3 / 128, (BT + 127) / 128), 256, 0, stream>>>(
      Ab, WaT, aqkv, nullptr, nullptr, BT, INNER3, DIM);

  // fused bidirectional attention -> concatenated bf16 activations
  k_attn<<<BATCH * NHEADS, 256, 0, stream>>>(vqkv, aqkv, mask, cat);

  // output projection: 41120 x 512 x 512 (+bias), f32 out
  k_gemm_bf16<<<dim3(DIM / 128, (MCAT + 127) / 128), 256, 0, stream>>>(
      cat, WoT, nullptr, out, bo, MCAT, DIM, DIM);
}
